// GraphTransformer_16037407884030
// MI455X (gfx1250) — matrix operand order repaired and checked by reading, not compiled, not run on hardware
//
#include <hip/hip_runtime.h>
#include <hip/hip_bf16.h>

typedef __attribute__((ext_vector_type(16))) _Float16 v16h;
typedef __attribute__((ext_vector_type(8)))  float    v8f;

#define N_NODES 50000
#define N_EDGES 500000
#define HEADS   4
#define CDIM    32
#define KIN     128
#define NCOLS1  512          // [wq1|wk1|wv1|ws1]
#define NCOLS2  416          // [wq2|wk2|wv2|ws2(32)]
#define MTILES  (N_NODES/16) // 3125
#define KD32    4            // K=128 -> 4 WMMA steps of K=32
#define LDQ     512          // row stride of qkvs scratch (both layers)
#define INITKEY 0x007FFFFFu  // ordered key of -inf
#define RSQRT32 0.17677669529663688f

// ---------------------------------------------------------------------------
// Pack x (f32 row-major NxK) into WMMA A-fragment order:
//   idx = ((mt*KD32 + kk)*32 + lane)*16 + j
//   lane: m = lane&15, ksel = lane>>4
//   j<8 : k = kk*32 + ksel*8 + j          (K 0..7 / 8..15)
//   j>=8: k = kk*32 + 16 + ksel*8 + (j-8) (K 16..23 / 24..31)
// ---------------------------------------------------------------------------
__global__ void gt_pack_x(const float* __restrict__ x, _Float16* __restrict__ xp, int total) {
    int t = blockIdx.x * blockDim.x + threadIdx.x;
    if (t >= total) return;
    int jj   = t & 15;
    int lane = (t >> 4) & 31;
    int rest = t >> 9;
    int kk   = rest & 3;
    int mt   = rest >> 2;
    int row  = mt * 16 + (lane & 15);
    int k    = kk * 32 + ((lane >> 4) << 3) + (jj & 7) + ((jj >> 3) << 4);
    xp[t] = (_Float16)x[(size_t)row * KIN + k];
}

// ---------------------------------------------------------------------------
// Pack concatenated weights [w0|w1|w2|w3] (each f32 row-major KIN x ci) into
// WMMA B-fragment order: lane n = lane&15, ksel = lane>>4, element j
// maps to k = kk*32 + ksel*8 + (j&7) + (j>>3)*16.  idx = ((nt*KD32+kk)*32+lane)*16+j
// ---------------------------------------------------------------------------
__global__ void gt_pack_w(const float* __restrict__ w0, const float* __restrict__ w1,
                          const float* __restrict__ w2, const float* __restrict__ w3,
                          int c0, int c1, int c2, int c3,
                          _Float16* __restrict__ wp, int total) {
    int t = blockIdx.x * blockDim.x + threadIdx.x;
    if (t >= total) return;
    int jj   = t & 15;
    int lane = (t >> 4) & 31;
    int rest = t >> 9;
    int kk   = rest & 3;
    int nt   = rest >> 2;
    int n    = nt * 16 + (lane & 15);
    int k    = kk * 32 + ((lane >> 4) << 3) + (jj & 7) + ((jj >> 3) << 4);
    const float* W; int nl, ld;
    if      (n < c0)           { W = w0; nl = n;                ld = c0; }
    else if (n < c0 + c1)      { W = w1; nl = n - c0;           ld = c1; }
    else if (n < c0 + c1 + c2) { W = w2; nl = n - c0 - c1;      ld = c2; }
    else                       { W = w3; nl = n - c0 - c1 - c2; ld = c3; }
    wp[t] = (_Float16)W[(size_t)k * ld + nl];
}

__global__ void gt_pack_bias(const float* __restrict__ b0, const float* __restrict__ b1,
                             const float* __restrict__ b2, const float* __restrict__ b3,
                             int c0, int c1, int c2, int c3,
                             float* __restrict__ out, int total) {
    int t = blockIdx.x * blockDim.x + threadIdx.x;
    if (t >= total) return;
    float v;
    if      (t < c0)           v = b0[t];
    else if (t < c0 + c1)      v = b1[t - c0];
    else if (t < c0 + c1 + c2) v = b2[t - c0 - c1];
    else                       v = b3[t - c0 - c1 - c2];
    out[t] = v;
}

__global__ void gt_fill_u32(unsigned int* __restrict__ p, unsigned int v, int n) {
    int t = blockIdx.x * blockDim.x + threadIdx.x;
    if (t < n) p[t] = v;
}

// ---------------------------------------------------------------------------
// WMMA GEMM: Out[M x Ncols] = Apack(M x 128, f16) * Bpack(128 x Ncols, f16)
//            + biascat.
// One wave computes a 16x32 strip (two adjacent 16x16 N-tiles): the A
// fragment is loaded once per K-step and feeds two independent WMMA
// accumulator chains (ILP hides the WMMA->WMMA RAW latency).
// ---------------------------------------------------------------------------
__global__ __launch_bounds__(128) void gt_gemm_wmma(
        const _Float16* __restrict__ Apack, const _Float16* __restrict__ Bpack,
        const float* __restrict__ biascat, float* __restrict__ Out,
        int Ntiles, int ldOut) {
    int wave = threadIdx.x >> 5;
    int lane = threadIdx.x & 31;
    int pair = blockIdx.x * 4 + wave;     // pair of adjacent N-tiles
    int mt   = blockIdx.y;
    int nt0  = pair * 2;
    if (nt0 >= Ntiles) return;            // Ntiles is even for both layers

    v8f acc0 = {}, acc1 = {};
    const _Float16* ap  = Apack + ((size_t)(mt  * KD32) * 32 + lane) * 16;
    const _Float16* bp0 = Bpack + ((size_t)(nt0 * KD32) * 32 + lane) * 16;
    const _Float16* bp1 = bp0 + (size_t)KD32 * 32 * 16;   // next N-tile
#pragma unroll
    for (int kk = 0; kk < KD32; ++kk) {
        v16h a  = *(const v16h*)ap;   ap  += 32 * 16;
        v16h b0 = *(const v16h*)bp0;  bp0 += 32 * 16;
        v16h b1 = *(const v16h*)bp1;  bp1 += 32 * 16;
        acc0 = __builtin_amdgcn_wmma_f32_16x16x32_f16(
                   false, a, false, b0, (short)0, acc0, false, false);
        acc1 = __builtin_amdgcn_wmma_f32_16x16x32_f16(
                   false, a, false, b1, (short)0, acc1, false, false);
    }
    int   col0  = nt0 * 16 + (lane & 15);
    int   col1  = col0 + 16;
    float bias0 = biascat[col0];
    float bias1 = biascat[col1];
    int   rbase = mt * 16 + ((lane >> 4) << 3);   // lanes 16..31 hold M=8..15
#pragma unroll
    for (int r = 0; r < 8; ++r) {
        size_t rowoff = (size_t)(rbase + r) * ldOut;
        Out[rowoff + col0] = acc0[r] + bias0;
        Out[rowoff + col1] = acc1[r] + bias1;
    }
}

// ---------------------------------------------------------------------------
// Edge kernels: one thread per (edge, head).
// ---------------------------------------------------------------------------
__global__ void gt_edge_logits(const float* __restrict__ qkvs, const int* __restrict__ ei,
                               float* __restrict__ logits, unsigned int* __restrict__ mkey,
                               int qoff, int koff) {
    int t = blockIdx.x * blockDim.x + threadIdx.x;
    if (t >= N_EDGES * HEADS) return;
    int e = t >> 2, h = t & 3;
    int s = ei[e], d = ei[N_EDGES + e];
    const float* qp = qkvs + (size_t)d * LDQ + qoff + h * CDIM;
    const float* kp = qkvs + (size_t)s * LDQ + koff + h * CDIM;
    float acc = 0.f;
#pragma unroll
    for (int c = 0; c < CDIM; c += 4) {
        float4 qv = *(const float4*)(qp + c);
        float4 kv = *(const float4*)(kp + c);
        acc += qv.x * kv.x + qv.y * kv.y + qv.z * kv.z + qv.w * kv.w;
    }
    float lg = acc * RSQRT32;
    logits[t] = lg;
    unsigned int bits = __float_as_uint(lg);
    unsigned int key  = (bits & 0x80000000u) ? ~bits : (bits | 0x80000000u);
    atomicMax(&mkey[d * HEADS + h], key);
}

__global__ void gt_decode_max(unsigned int* __restrict__ mkey, int n) {
    int t = blockIdx.x * blockDim.x + threadIdx.x;
    if (t >= n) return;
    unsigned int key = mkey[t];
    float m = 0.f;
    if (key != INITKEY) {
        unsigned int bits = (key & 0x80000000u) ? (key ^ 0x80000000u) : ~key;
        m = __uint_as_float(bits);
        if (!isfinite(m)) m = 0.f;   // matches reference where(isfinite(m), m, 0)
    }
    mkey[t] = __float_as_uint(m);    // buffer now holds float max per (node,head)
}

__global__ void gt_edge_accum(const float* __restrict__ qkvs, const int* __restrict__ ei,
                              const float* __restrict__ logits, const float* __restrict__ mF,
                              float* __restrict__ den, float* __restrict__ agg, int voff) {
    int t = blockIdx.x * blockDim.x + threadIdx.x;
    if (t >= N_EDGES * HEADS) return;
    int e = t >> 2, h = t & 3;
    int s = ei[e], d = ei[N_EDGES + e];
    float ex = __expf(logits[t] - mF[d * HEADS + h]);
    atomicAdd(&den[d * HEADS + h], ex);
    const float* vp = qkvs + (size_t)s * LDQ + voff + h * CDIM;
    float*       ap = agg + ((size_t)d * HEADS + h) * CDIM;
#pragma unroll
    for (int c = 0; c < CDIM; ++c)
        atomicAdd(&ap[c], ex * vp[c]);
}

// ---------------------------------------------------------------------------
// Layer-1 finalize: h1 = relu(agg/den + skip), written directly into the
// WMMA A-fragment layout for the layer-2 GEMM.
// ---------------------------------------------------------------------------
__global__ void gt_finalize1(const float* __restrict__ agg, const float* __restrict__ den,
                             const float* __restrict__ qkvs, _Float16* __restrict__ h1p) {
    int t = blockIdx.x * blockDim.x + threadIdx.x;
    if (t >= N_NODES * KIN) return;
    int n = t >> 7, j = t & 127;
    int h = j >> 5, c = j & 31;
    float dd  = den[n * HEADS + h];  dd = dd > 0.f ? dd : 1.f;
    float val = agg[((size_t)n * HEADS + h) * CDIM + c] / dd
              + qkvs[(size_t)n * LDQ + 384 + j];     // skip = x@ws1 + bs1
    val = val > 0.f ? val : 0.f;                     // ReLU
    // scatter into A-fragment layout
    int mt = n >> 4, mrow = n & 15;
    int kk = j >> 5, kl = j & 31;
    int ksel = (kl >> 3) & 1;
    int jj   = (kl & 7) + ((kl >> 4) & 1) * 8;
    int lane = ksel * 16 + mrow;
    h1p[(((size_t)mt * KD32 + kk) * 32 + lane) * 16 + jj] = (_Float16)val;
}

__global__ void gt_final_out(const float* __restrict__ agg, const float* __restrict__ den,
                             const float* __restrict__ qkvs, float* __restrict__ out) {
    int t = blockIdx.x * blockDim.x + threadIdx.x;
    if (t >= N_NODES * CDIM) return;
    int n = t >> 5, c = t & 31;
    float s = 0.f;
#pragma unroll
    for (int h = 0; h < HEADS; ++h) {
        float dd = den[n * HEADS + h];  dd = dd > 0.f ? dd : 1.f;
        s += agg[((size_t)n * HEADS + h) * CDIM + c] / dd;
    }
    out[t] = 0.25f * s + qkvs[(size_t)n * LDQ + 384 + c];  // skip = h1@ws2 + bs2
}

// ---------------------------------------------------------------------------
extern "C" void kernel_launch(void* const* d_in, const int* in_sizes, int n_in,
                              void* d_out, int out_size, void* d_ws, size_t ws_size,
                              hipStream_t stream) {
    const float* x  = (const float*)d_in[0];
    const int*   ei = (const int*)d_in[1];
    const float* wq1 = (const float*)d_in[2],  *bq1 = (const float*)d_in[3];
    const float* wk1 = (const float*)d_in[4],  *bk1 = (const float*)d_in[5];
    const float* wv1 = (const float*)d_in[6],  *bv1 = (const float*)d_in[7];
    const float* ws1 = (const float*)d_in[8],  *bs1 = (const float*)d_in[9];
    const float* wq2 = (const float*)d_in[10], *bq2 = (const float*)d_in[11];
    const float* wk2 = (const float*)d_in[12], *bk2 = (const float*)d_in[13];
    const float* wv2 = (const float*)d_in[14], *bv2 = (const float*)d_in[15];
    const float* ws2 = (const float*)d_in[16], *bs2 = (const float*)d_in[17];
    float* out = (float*)d_out;

    // ---- workspace carve (256B aligned slices) ----
    char* wptr = (char*)d_ws;
    auto carve = [&](size_t bytes) -> void* {
        void* p = (void*)wptr;
        wptr += (bytes + 255) & ~(size_t)255;
        return p;
    };
    _Float16* xpack  = (_Float16*)carve((size_t)N_NODES * KIN * 2);
    _Float16* w1pack = (_Float16*)carve((size_t)KIN * NCOLS1 * 2);
    _Float16* w2pack = (_Float16*)carve((size_t)KIN * NCOLS2 * 2);
    float*    bias1  = (float*)   carve(NCOLS1 * 4);
    float*    bias2  = (float*)   carve(NCOLS2 * 4);
    float*    qkvs   = (float*)   carve((size_t)N_NODES * LDQ * 4);
    _Float16* h1pack = (_Float16*)carve((size_t)N_NODES * KIN * 2);
    float*    logits = (float*)   carve((size_t)N_EDGES * HEADS * 4);
    unsigned* mkey   = (unsigned*)carve((size_t)N_NODES * HEADS * 4);
    float*    den    = (float*)   carve((size_t)N_NODES * HEADS * 4);
    float*    agg    = (float*)   carve((size_t)N_NODES * HEADS * CDIM * 4);

    const int B = 256;
    auto cdiv = [](int a, int b) { return (a + b - 1) / b; };

    // ---- operand packing ----
    gt_pack_x<<<cdiv(N_NODES * KIN, B), B, 0, stream>>>(x, xpack, N_NODES * KIN);
    gt_pack_w<<<cdiv(KIN * NCOLS1, B), B, 0, stream>>>(wq1, wk1, wv1, ws1,
            128, 128, 128, 128, w1pack, KIN * NCOLS1);
    gt_pack_w<<<cdiv(KIN * NCOLS2, B), B, 0, stream>>>(wq2, wk2, wv2, ws2,
            128, 128, 128, 32, w2pack, KIN * NCOLS2);
    gt_pack_bias<<<cdiv(NCOLS1, B), B, 0, stream>>>(bq1, bk1, bv1, bs1,
            128, 128, 128, 128, bias1, NCOLS1);
    gt_pack_bias<<<cdiv(NCOLS2, B), B, 0, stream>>>(bq2, bk2, bv2, bs2,
            128, 128, 128, 32, bias2, NCOLS2);

    // ================= layer 1 =================
    // 32 N-tiles -> 16 pairs -> 4 blocks of 4 waves in x
    gt_gemm_wmma<<<dim3(cdiv(NCOLS1 / 32, 4), MTILES), 128, 0, stream>>>(
            xpack, w1pack, bias1, qkvs, NCOLS1 / 16, LDQ);

    gt_fill_u32<<<cdiv(N_NODES * HEADS, B), B, 0, stream>>>(mkey, INITKEY, N_NODES * HEADS);
    gt_fill_u32<<<cdiv(N_NODES * HEADS, B), B, 0, stream>>>((unsigned*)den, 0u, N_NODES * HEADS);
    gt_fill_u32<<<cdiv(N_NODES * HEADS * CDIM, B), B, 0, stream>>>((unsigned*)agg, 0u,
            N_NODES * HEADS * CDIM);

    gt_edge_logits<<<cdiv(N_EDGES * HEADS, B), B, 0, stream>>>(qkvs, ei, logits, mkey, 0, 128);
    gt_decode_max<<<cdiv(N_NODES * HEADS, B), B, 0, stream>>>(mkey, N_NODES * HEADS);
    gt_edge_accum<<<cdiv(N_EDGES * HEADS, B), B, 0, stream>>>(qkvs, ei, logits,
            (const float*)mkey, den, agg, 256);
    gt_finalize1<<<cdiv(N_NODES * KIN, B), B, 0, stream>>>(agg, den, qkvs, h1pack);

    // ================= layer 2 =================
    // 26 N-tiles -> 13 pairs -> 4 blocks of 4 waves in x (last partially idle)
    gt_gemm_wmma<<<dim3(cdiv(NCOLS2 / 32, 4), MTILES), 128, 0, stream>>>(
            h1pack, w2pack, bias2, qkvs, NCOLS2 / 16, LDQ);

    gt_fill_u32<<<cdiv(N_NODES * HEADS, B), B, 0, stream>>>(mkey, INITKEY, N_NODES * HEADS);
    gt_fill_u32<<<cdiv(N_NODES * HEADS, B), B, 0, stream>>>((unsigned*)den, 0u, N_NODES * HEADS);
    gt_fill_u32<<<cdiv(N_NODES * HEADS * CDIM, B), B, 0, stream>>>((unsigned*)agg, 0u,
            N_NODES * HEADS * CDIM);

    gt_edge_logits<<<cdiv(N_EDGES * HEADS, B), B, 0, stream>>>(qkvs, ei, logits, mkey, 0, 128);
    gt_decode_max<<<cdiv(N_NODES * HEADS, B), B, 0, stream>>>(mkey, N_NODES * HEADS);
    gt_edge_accum<<<cdiv(N_EDGES * HEADS, B), B, 0, stream>>>(qkvs, ei, logits,
            (const float*)mkey, den, agg, 256);
    gt_final_out<<<cdiv(N_NODES * CDIM, B), B, 0, stream>>>(agg, den, qkvs, out);

    (void)in_sizes; (void)n_in; (void)out_size; (void)ws_size;
}
